// SABiLSTMModel_64183991272078
// MI455X (gfx1250) — compile-verified, matrix operand order checked by reading, not run on hardware
//
#include <hip/hip_runtime.h>
#include <math.h>

// ---------------------------------------------------------------------------
// SABiLSTM for MI455X (gfx1250): bf16 WMMA for the parallel GEMMs,
// persistent wave32 workgroups for the sequential LSTM recurrence.
// ---------------------------------------------------------------------------

typedef __attribute__((ext_vector_type(16))) __bf16 v16bf;
typedef __attribute__((ext_vector_type(8)))  float  v8f;

static constexpr int T    = 16384;
static constexpr int D    = 300;
static constexpr int U    = 300;
static constexpr int G4   = 1200;  // 4*U gate width
static constexpr int ADIM = 350;
static constexpr int R    = 30;
static constexpr int H2   = 600;   // 2*U

static constexpr int KPX  = 320;   // D padded to mult of 32 (WMMA K)
static constexpr int NPG  = 1216;  // G4 padded to mult of 64 (tile N)
static constexpr int KPH  = 608;   // H2 padded to mult of 32
static constexpr int NPA  = 384;   // ADIM padded to mult of 64

// ---- bf16 helpers ---------------------------------------------------------
__device__ __forceinline__ __bf16 f2bf(float f) {
  unsigned u = __builtin_bit_cast(unsigned, f);
  unsigned r = (u + 0x7fffu + ((u >> 16) & 1u)) >> 16;
  unsigned short s = (unsigned short)r;
  return __builtin_bit_cast(__bf16, s);
}
// native cast so the backend can select v_fma_mix_f32_bf16 in fma chains
__device__ __forceinline__ float bf2f(__bf16 b) { return (float)b; }

__device__ __forceinline__ float sigmoidf_(float x) {
  return 1.0f / (1.0f + expf(-x));
}

// K-pair base index for the 16-bit A/B WMMA fragment layout
// (ISA 7.12.2: VGPR r holds packed K={k0,k0+1}; lanes 16-31 shift K by 8).
__device__ __forceinline__ int frag_k0(int r, int half) {
  return ((r < 4) ? (2 * r) : (16 + 2 * (r - 4))) + 8 * half;
}

// ---- convert fp32 -> bf16 with zero padding -------------------------------
__global__ void k_cvt_pad(const float* __restrict__ src, __bf16* __restrict__ dst,
                          int srcRows, int srcCols, int dstRows, int dstCols) {
  size_t total = (size_t)dstRows * dstCols;
  for (size_t i = (size_t)blockIdx.x * blockDim.x + threadIdx.x; i < total;
       i += (size_t)gridDim.x * blockDim.x) {
    int r = (int)(i / dstCols);
    int c = (int)(i % dstCols);
    float v = (r < srcRows && c < srcCols) ? src[(size_t)r * srcCols + c] : 0.0f;
    dst[i] = f2bf(v);
  }
}

// transpose-convert: dst[k][n] = src[n][k]  (for W1: (350,600) -> (608,384))
__global__ void k_cvtT_pad(const float* __restrict__ src, __bf16* __restrict__ dst,
                           int srcRows, int srcCols, int dstRows, int dstCols) {
  size_t total = (size_t)dstRows * dstCols;
  for (size_t i = (size_t)blockIdx.x * blockDim.x + threadIdx.x; i < total;
       i += (size_t)gridDim.x * blockDim.x) {
    int k = (int)(i / dstCols);
    int n = (int)(i % dstCols);
    float v = (n < srcRows && k < srcCols) ? src[(size_t)n * srcCols + k] : 0.0f;
    dst[i] = f2bf(v);
  }
}

// ---- WMMA bf16 GEMM: C[MxN] = A[MxKp] * B[KpxN] (+bias) (opt tanh) --------
// Macro-tile 256x64, 8 waves, each wave owns a 32x64 strip:
// 2 A-fragments x 4 B-fragments = 8 v_wmma_f32_16x16x32_bf16 per K-step.
// A staged row-major, B staged TRANSPOSED (k-contiguous per column) so both
// fragment builds collapse into two 16B-aligned ds_load_b128 per fragment.
// Row stride 40 bf16 (80 B): keeps 16B alignment, bank-conflict free.
#define BM 256
#define BN 64
#define BK 32
#define LDT 40   // LDS row stride in bf16 elements for both tiles

__global__ __launch_bounds__(256)
void wmma_gemm_bf16(const __bf16* __restrict__ A, const __bf16* __restrict__ B,
                    float* __restrict__ C, const float* __restrict__ bias,
                    int M, int Kp, int N, int ldc, int biasN, int do_tanh) {
  __shared__ __align__(16) __bf16 Alds[BM * LDT];  // 20.0 KB
  __shared__ __align__(16) __bf16 Blds[BN * LDT];  //  5.0 KB

  const int tid  = threadIdx.x;
  const int wave = tid >> 5;   // 0..7 -> rows wave*32 .. wave*32+31
  const int lane = tid & 31;
  const int row0 = blockIdx.x * BM;
  const int col0 = blockIdx.y * BN;

  const int m    = lane & 15;
  const int half = lane >> 4;

  v8f acc[2][4];
#pragma unroll
  for (int mi = 0; mi < 2; ++mi)
#pragma unroll
    for (int j = 0; j < 4; ++j)
#pragma unroll
      for (int r = 0; r < 8; ++r) acc[mi][j][r] = 0.0f;

  const int Kd = Kp >> 1;  // dwords per A row
  const int Nd = N  >> 1;  // dwords per B row
  const unsigned* Ag = (const unsigned*)A;
  const unsigned* Bg = (const unsigned*)B;

  // register double-buffer for the global tiles
  unsigned aReg[16];  // BM*BK/2 / 256 = 16 dwords per thread
  unsigned bReg[4];   // BK*BN/2 / 256 = 4 dwords per thread

  auto fetch = [&](int kp) {
#pragma unroll
    for (int jj = 0; jj < 16; ++jj) {
      int idx = tid + jj * 256;
      int r = idx >> 4;        // 16 dwords per A row
      int d = idx & 15;
      aReg[jj] = Ag[(size_t)(row0 + r) * Kd + (kp >> 1) + d];
    }
#pragma unroll
    for (int jj = 0; jj < 4; ++jj) {
      int idx = tid + jj * 256;
      int k = idx >> 5;        // 32 dwords per B row
      int d = idx & 31;        // n-pair index
      bReg[jj] = Bg[(size_t)(kp + k) * Nd + (col0 >> 1) + d];
    }
  };
  auto stage = [&]() {
#pragma unroll
    for (int jj = 0; jj < 16; ++jj) {
      int idx = tid + jj * 256;
      int r = idx >> 4;
      int d = idx & 15;
      *(unsigned*)&Alds[r * LDT + 2 * d] = aReg[jj];  // LDT even -> aligned
    }
#pragma unroll
    for (int jj = 0; jj < 4; ++jj) {
      int idx = tid + jj * 256;
      int k = idx >> 5;
      int d = idx & 31;
      unsigned v = bReg[jj];
      Blds[(2 * d    ) * LDT + k] = __builtin_bit_cast(__bf16, (unsigned short)(v & 0xffffu));
      Blds[(2 * d + 1) * LDT + k] = __builtin_bit_cast(__bf16, (unsigned short)(v >> 16));
    }
  };

  fetch(0);
  for (int kp = 0; kp < Kp; kp += BK) {
    stage();
    __syncthreads();
    if (kp + BK < Kp) fetch(kp + BK);  // overlap next global tile with math

    v16bf afrag[2];
#pragma unroll
    for (int mi = 0; mi < 2; ++mi)
#pragma unroll
      for (int r = 0; r < 8; ++r) {
        int k0 = frag_k0(r, half);
        const __bf16* ap = &Alds[(wave * 32 + mi * 16 + m) * LDT + k0];
        afrag[mi][2 * r]     = ap[0];
        afrag[mi][2 * r + 1] = ap[1];
      }
#pragma unroll
    for (int j = 0; j < 4; ++j) {
      v16bf bfrag;
#pragma unroll
      for (int r = 0; r < 8; ++r) {
        int k0 = frag_k0(r, half);
        const __bf16* bp = &Blds[(j * 16 + m) * LDT + k0];
        bfrag[2 * r]     = bp[0];
        bfrag[2 * r + 1] = bp[1];
      }
#pragma unroll
      for (int mi = 0; mi < 2; ++mi)
        acc[mi][j] = __builtin_amdgcn_wmma_f32_16x16x32_bf16(
            false, afrag[mi], false, bfrag, (short)0, acc[mi][j], false, false);
    }
    __syncthreads();
  }

  // epilogue: C layout — VGPR r holds M = r + 8*half, N = lane&15
#pragma unroll
  for (int mi = 0; mi < 2; ++mi)
#pragma unroll
    for (int j = 0; j < 4; ++j)
#pragma unroll
      for (int r = 0; r < 8; ++r) {
        int row = row0 + wave * 32 + mi * 16 + r + 8 * half;
        int col = col0 + j * 16 + m;
        float v = acc[mi][j][r];
        if (bias) v += (col < biasN) ? bias[col] : 0.0f;
        if (do_tanh) v = tanhf(v);
        C[(size_t)row * ldc + col] = v;
      }
}

// ---- sequential LSTM: one persistent block per direction ------------------
__global__ __launch_bounds__(1024)
void k_lstm(const float* __restrict__ xp_f, const float* __restrict__ xp_b,
            const __bf16* __restrict__ Wr_f, const __bf16* __restrict__ Wr_b,
            float* __restrict__ H) {
  const int dir = blockIdx.x;               // 0 = forward, 1 = backward
  const float*  xp = dir ? xp_b : xp_f;     // [T][NPG]
  const __bf16* Wr = dir ? Wr_b : Wr_f;     // [U][G4] bf16, L2-resident

  __shared__ float hs[U];
  __shared__ float cs[U];
  __shared__ float zs[G4];

  const int tid = threadIdx.x;
  for (int i = tid; i < U; i += blockDim.x) { hs[i] = 0.0f; cs[i] = 0.0f; }
  __syncthreads();

  for (int step = 0; step < T; ++step) {
    const int t  = dir ? (T - 1 - step) : step;
    const int tn = dir ? (t > 0 ? t - 1 : t) : (t + 1 < T ? t + 1 : t);
    if (tid < G4) __builtin_prefetch(&xp[(size_t)tn * NPG + tid], 0, 1);

    // z = sigmoid(xp[t] + h @ Wr); fma_mix from bf16 weights
    for (int col = tid; col < G4; col += blockDim.x) {
      float acc = xp[(size_t)t * NPG + col];
      const __bf16* wc = Wr + col;
#pragma unroll 8
      for (int u = 0; u < U; ++u)
        acc = fmaf(hs[u], bf2f(wc[(size_t)u * G4]), acc);
      zs[col] = sigmoidf_(acc);
    }
    __syncthreads();

    if (tid < U) {
      float i_ = zs[tid];
      float f_ = zs[tid + U];
      float g_ = zs[tid + 2 * U];
      float o_ = zs[tid + 3 * U];
      float c  = f_ * cs[tid] + i_ * g_;
      float h  = o_ * sigmoidf_(c);
      cs[tid] = c;
      hs[tid] = h;
      H[(size_t)t * H2 + dir * U + tid] = h;
    }
    __syncthreads();
  }
}

// ---- G[r][t] = sum_a W2[r][a] * S[t][a] (S already tanh'd) ----------------
__global__ __launch_bounds__(256)
void k_att_gemm2(const float* __restrict__ S, const float* __restrict__ W2,
                 float* __restrict__ G) {
  __shared__ float w2s[R * ADIM];  // 42 KB
  for (int i = threadIdx.x; i < R * ADIM; i += 256) w2s[i] = W2[i];
  __syncthreads();

  const int t = blockIdx.x * 256 + threadIdx.x;
  float acc[R];
#pragma unroll
  for (int r = 0; r < R; ++r) acc[r] = 0.0f;

  for (int a = 0; a < ADIM; ++a) {
    float s = S[(size_t)t * NPA + a];
#pragma unroll
    for (int r = 0; r < R; ++r) acc[r] = fmaf(w2s[r * ADIM + a], s, acc[r]);
  }
#pragma unroll
  for (int r = 0; r < R; ++r) G[(size_t)r * T + t] = acc[r];
}

// ---- row softmax over T=16384 ---------------------------------------------
__global__ __launch_bounds__(256)
void k_softmax_row(const float* __restrict__ G, float* __restrict__ A) {
  __shared__ float red[256];
  const int r = blockIdx.x;
  const int tid = threadIdx.x;
  const float* g = G + (size_t)r * T;

  float m = -3.4e38f;
  for (int t = tid; t < T; t += 256) m = fmaxf(m, g[t]);
  red[tid] = m; __syncthreads();
  for (int s = 128; s > 0; s >>= 1) {
    if (tid < s) red[tid] = fmaxf(red[tid], red[tid + s]);
    __syncthreads();
  }
  m = red[0]; __syncthreads();

  float sum = 0.0f;
  for (int t = tid; t < T; t += 256) sum += expf(g[t] - m);
  red[tid] = sum; __syncthreads();
  for (int s = 128; s > 0; s >>= 1) {
    if (tid < s) red[tid] += red[tid + s];
    __syncthreads();
  }
  const float inv = 1.0f / red[0];
  for (int t = tid; t < T; t += 256)
    A[(size_t)r * T + t] = expf(g[t] - m) * inv;
}

// ---- Ma[r][j] = sum_t A[r][t] * H[t][j] -----------------------------------
__global__ __launch_bounds__(256)
void k_pool(const float* __restrict__ A, const float* __restrict__ H,
            float* __restrict__ Ma) {
  __shared__ float red[256];
  const int j = blockIdx.x;   // 0..599
  const int r = blockIdx.y;   // 0..29
  const int tid = threadIdx.x;

  float acc = 0.0f;
  for (int t = tid; t < T; t += 256)
    acc = fmaf(A[(size_t)r * T + t], H[(size_t)t * H2 + j], acc);
  red[tid] = acc; __syncthreads();
  for (int s = 128; s > 0; s >>= 1) {
    if (tid < s) red[tid] += red[tid + s];
    __syncthreads();
  }
  if (tid == 0) Ma[(size_t)r * H2 + j] = red[0];
}

// ---- out = sigmoid(mean_r(Ma @ dense_w) + dense_b) ------------------------
__global__ __launch_bounds__(256)
void k_final(const float* __restrict__ Ma, const float* __restrict__ w,
             const float* __restrict__ b, float* __restrict__ out) {
  __shared__ float red[256];
  const int tid = threadIdx.x;
  float acc = 0.0f;
  for (int i = tid; i < R * H2; i += 256)
    acc = fmaf(Ma[i], w[i % H2], acc);
  red[tid] = acc; __syncthreads();
  for (int s = 128; s > 0; s >>= 1) {
    if (tid < s) red[tid] += red[tid + s];
    __syncthreads();
  }
  if (tid == 0) out[0] = sigmoidf_(red[0] / (float)R + b[0]);
}

// ---------------------------------------------------------------------------
extern "C" void kernel_launch(void* const* d_in, const int* in_sizes, int n_in,
                              void* d_out, int out_size, void* d_ws, size_t ws_size,
                              hipStream_t stream) {
  (void)in_sizes; (void)n_in; (void)out_size; (void)ws_size;
  const float* emb  = (const float*)d_in[0];
  const float* Wk_f = (const float*)d_in[1];
  const float* Wr_f = (const float*)d_in[2];
  const float* b_f  = (const float*)d_in[3];
  const float* Wk_b = (const float*)d_in[4];
  const float* Wr_b = (const float*)d_in[5];
  const float* b_b  = (const float*)d_in[6];
  const float* W1   = (const float*)d_in[7];
  const float* W2   = (const float*)d_in[8];
  const float* dw   = (const float*)d_in[9];
  const float* db   = (const float*)d_in[10];
  float* out = (float*)d_out;

  char* base = (char*)d_ws;
  size_t cur = 0;
  auto carve = [&](size_t bytes) -> char* {
    char* p = base + cur;
    cur += (bytes + 255) & ~(size_t)255;
    return p;
  };
  __bf16* xbf  = (__bf16*)carve((size_t)T * KPX * 2);    // x padded
  __bf16* wkf  = (__bf16*)carve((size_t)KPX * NPG * 2);  // Wk_f padded
  __bf16* wkb  = (__bf16*)carve((size_t)KPX * NPG * 2);
  __bf16* wrf  = (__bf16*)carve((size_t)U * G4 * 2);     // Wr_f bf16
  __bf16* wrb  = (__bf16*)carve((size_t)U * G4 * 2);
  __bf16* w1t  = (__bf16*)carve((size_t)KPH * NPA * 2);  // W1^T padded
  float*  xpf  = (float*)carve((size_t)T * NPG * 4);     // gate pre-acts fwd
  float*  xpb  = (float*)carve((size_t)T * NPG * 4);
  float*  Hbuf = (float*)carve((size_t)T * H2 * 4);      // [hf|hb]
  __bf16* hbf  = (__bf16*)carve((size_t)T * KPH * 2);    // H padded bf16
  float*  Sbuf = (float*)carve((size_t)T * NPA * 4);     // tanh(H @ W1^T)
  float*  Gbuf = (float*)carve((size_t)R * T * 4);
  float*  Abuf = (float*)carve((size_t)R * T * 4);
  float*  Mbuf = (float*)carve((size_t)R * H2 * 4);

  auto grid1 = [](size_t total) { return dim3((unsigned)((total + 255) / 256)); };

  // stage 0: precision conversion + padding
  k_cvt_pad <<<grid1((size_t)T * KPX),   256, 0, stream>>>(emb,  xbf, T,    D,  T,   KPX);
  k_cvt_pad <<<grid1((size_t)KPX * NPG), 256, 0, stream>>>(Wk_f, wkf, D,    G4, KPX, NPG);
  k_cvt_pad <<<grid1((size_t)KPX * NPG), 256, 0, stream>>>(Wk_b, wkb, D,    G4, KPX, NPG);
  k_cvt_pad <<<grid1((size_t)U * G4),    256, 0, stream>>>(Wr_f, wrf, U,    G4, U,   G4);
  k_cvt_pad <<<grid1((size_t)U * G4),    256, 0, stream>>>(Wr_b, wrb, U,    G4, U,   G4);
  k_cvtT_pad<<<grid1((size_t)KPH * NPA), 256, 0, stream>>>(W1,   w1t, ADIM, H2, KPH, NPA);

  // stage 1: gate pre-activations xp = x @ Wk + b (WMMA bf16)
  dim3 gxp(T / BM, NPG / BN);
  wmma_gemm_bf16<<<gxp, 256, 0, stream>>>(xbf, wkf, xpf, b_f, T, KPX, NPG, NPG, G4, 0);
  wmma_gemm_bf16<<<gxp, 256, 0, stream>>>(xbf, wkb, xpb, b_b, T, KPX, NPG, NPG, G4, 0);

  // stage 2: sequential bi-directional recurrence (persistent blocks)
  k_lstm<<<2, 1024, 0, stream>>>(xpf, xpb, wrf, wrb, Hbuf);

  // stage 3: attention scores S = tanh(H @ W1^T) (WMMA bf16, fused tanh)
  k_cvt_pad<<<grid1((size_t)T * KPH), 256, 0, stream>>>(Hbuf, hbf, T, H2, T, KPH);
  dim3 gat(T / BM, NPA / BN);
  wmma_gemm_bf16<<<gat, 256, 0, stream>>>(hbf, w1t, Sbuf, nullptr, T, KPH, NPA, NPA, 0, 1);

  // stage 4: G = W2 * S^T, softmax over T, pool, head
  k_att_gemm2 <<<T / 256, 256, 0, stream>>>(Sbuf, W2, Gbuf);
  k_softmax_row<<<R, 256, 0, stream>>>(Gbuf, Abuf);
  k_pool<<<dim3(H2, R), 256, 0, stream>>>(Abuf, Hbuf, Mbuf);
  k_final<<<1, 256, 0, stream>>>(Mbuf, dw, db, out);
}
